// SE3NeuralFlows_78735340470909
// MI455X (gfx1250) — matrix operand order repaired and checked by reading, not compiled, not run on hardware
//
#include <hip/hip_runtime.h>
#include <math.h>

typedef __attribute__((ext_vector_type(16))) _Float16 v16h;
typedef __attribute__((ext_vector_type(8)))  float    v8f;

constexpr int DIMC   = 6;
constexpr int HALFC  = 3;
constexpr int KC     = 40;
constexpr int HC     = 128;   // hidden width
constexpr int HKC    = 120;   // HALF*K
constexpr int NLC    = 8;
constexpr int BLK_S  = 64;    // samples per block
constexpr int NTHR   = 128;   // 4 waves of 32
constexpr float PI_F = 3.14159265358979323846f;

union FragH {
    v16h     v;
    unsigned u[8];
};

// ---------------------------------------------------------------------------
// Prep: convert W2 (NL,128,128) and W3 (NL,128,120) to f16, column-major
// (n-major: Wt[i][n][k] = W[i][k][n]), W3 zero-padded to 128 output cols.
// ---------------------------------------------------------------------------
__global__ void se3flow_prep_weights(const float* __restrict__ W2,
                                     const float* __restrict__ W3,
                                     _Float16* __restrict__ W2t,
                                     _Float16* __restrict__ W3t) {
    int idx = blockIdx.x * blockDim.x + threadIdx.x;
    int total = NLC * HC * HC;
    if (idx >= total) return;
    int i = idx / (HC * HC);
    int r = idx % (HC * HC);
    int n = r / HC;              // output column of original W
    int k = r % HC;              // input row of original W
    W2t[idx] = (_Float16)W2[(i * HC + k) * HC + n];
    float w3 = (n < HKC) ? W3[(i * HC + k) * HKC + n] : 0.0f;
    W3t[idx] = (_Float16)w3;
}

// ---------------------------------------------------------------------------
// Main fused flow kernel: 128 threads (4 wave32), 64 samples per block.
// Per layer: { h1 on VALU || stage W2+W3 (64KB) into LDS },
// then GEMM2 and GEMM3 back-to-back (no barrier between: producer ==
// consumer wave, dscnt ordering), then spline + scatter. 4 barriers/layer.
// ---------------------------------------------------------------------------
__global__ __launch_bounds__(NTHR)
void se3flow_kernel(const float* __restrict__ x,
                    const float* __restrict__ W1,
                    const float* __restrict__ b1,
                    const float* __restrict__ b2,
                    const float* __restrict__ b3,
                    const int*   __restrict__ orders,
                    const int*   __restrict__ perms,
                    const _Float16* __restrict__ W2t,
                    const _Float16* __restrict__ W3t,
                    float* __restrict__ out) {
    __shared__ __align__(16) _Float16 bufA[BLK_S * HC];   // h1, later theta
    __shared__ __align__(16) _Float16 bufB[BLK_S * HC];   // h2
    __shared__ __align__(16) _Float16 sW2[HC * HC];       // staged W2t slice
    __shared__ __align__(16) _Float16 sW3[HC * HC];       // staged W3t slice
    __shared__ float    yblk[BLK_S * DIMC];
    __shared__ float    xtb [BLK_S * HALFC];
    __shared__ float    ytb [BLK_S * HALFC];

    const int tid  = threadIdx.x;
    const int lane = tid & 31;
    const int wave = tid >> 5;
    const int s0   = blockIdx.x * BLK_S;

    // ---- preprocess: x_p = clip, x_c = sphere2cube -------------------------
    if (tid < BLK_S) {
        const int s = s0 + tid;
        float v[DIMC];
#pragma unroll
        for (int d = 0; d < DIMC; ++d) v[d] = x[s * DIMC + d];
        float xs0 = v[3] / PI_F, xs1 = v[4] / PI_F, xs2 = v[5] / PI_F;
        float norm2 = xs0 * xs0 + xs1 * xs1 + xs2 * xs2;
        float den   = rsqrtf(fmaxf(1.0f - norm2, 1e-12f));
        float scl   = 2.0f / PI_F;
        float c0 = atanf(xs0 * den) * scl;
        float c1 = atanf(xs1 * den) * scl;
        float c2 = atanf(xs2 * den) * scl;
        bool inside = norm2 < 1.0f;
        yblk[tid * DIMC + 0] = fminf(fmaxf(v[0], -1.0f), 1.0f);
        yblk[tid * DIMC + 1] = fminf(fmaxf(v[1], -1.0f), 1.0f);
        yblk[tid * DIMC + 2] = fminf(fmaxf(v[2], -1.0f), 1.0f);
        yblk[tid * DIMC + 3] = inside ? c0 : 0.0f;
        yblk[tid * DIMC + 4] = inside ? c1 : 0.0f;
        yblk[tid * DIMC + 5] = inside ? c2 : 0.0f;
    }
    __syncthreads();

    for (int layer = 0; layer < NLC; ++layer) {
        // ---- per-layer uniform metadata (scalarizes to SMEM) ---------------
        int ord[DIMC], prm[DIMC];
#pragma unroll
        for (int d = 0; d < DIMC; ++d) {
            ord[d] = orders[layer * DIMC + d];
            prm[d] = (layer < NLC - 1) ? perms[layer * DIMC + d] : d;
        }

        // ---- Phase A: h1 on VALU  ||  stage W2+W3 (64KB) into LDS ----------
        {
            // weight copy: 4096 uint4 across 128 threads (loads overlap h1)
            const uint4* s2 = (const uint4*)(W2t + (size_t)layer * HC * HC);
            const uint4* s3 = (const uint4*)(W3t + (size_t)layer * HC * HC);
            uint4* d2 = (uint4*)sW2;
            uint4* d3 = (uint4*)sW3;
#pragma unroll
            for (int i = 0; i < (HC * HC / 8) / NTHR; ++i) {
                d2[tid + i * NTHR] = s2[tid + i * NTHR];
                d3[tid + i * NTHR] = s3[tid + i * NTHR];
            }

            // h1 = relu(xid @ W1 + b1): all 128 threads, 64 outputs each.
            // W1/b1 addresses are wave-uniform -> scalar (SMEM) loads.
            const int sH = tid & (BLK_S - 1);          // sample
            const int jb = (tid >> 6) * (HC / 2);      // 0 or 64 (wave-uniform)
            const float* w1l = W1 + (size_t)layer * HALFC * HC;
            const float* b1l = b1 + (size_t)layer * HC;
            float xi0 = yblk[sH * DIMC + ord[0]];
            float xi1 = yblk[sH * DIMC + ord[1]];
            float xi2 = yblk[sH * DIMC + ord[2]];
            if (tid < BLK_S) {
                xtb[tid * HALFC + 0] = yblk[tid * DIMC + ord[3]];
                xtb[tid * HALFC + 1] = yblk[tid * DIMC + ord[4]];
                xtb[tid * HALFC + 2] = yblk[tid * DIMC + ord[5]];
            }
            for (int j = jb; j < jb + HC / 2; ++j) {
                float a = b1l[j];
                a = fmaf(xi0, w1l[j], a);
                a = fmaf(xi1, w1l[HC + j], a);
                a = fmaf(xi2, w1l[2 * HC + j], a);
                bufA[sH * HC + j] = (_Float16)fmaxf(a, 0.0f);
            }
        }
        __syncthreads();

        // ---- WMMA fragment geometry (shared by both GEMMs) -----------------
        const int m0    = wave * 16;
        const int mrow  = m0 + (lane & 15);
        const int kgrpA = (lane >> 4) * 8;     // A: halves 0-7 -> kb.., 8-15 -> kb+16..
        const int n15   = lane & 15;
        const int kgrpB = (lane >> 4) * 8;     // B: halves 0-7 -> kb.., 8-15 -> kb+16..
        const int mbase = m0 + ((lane >> 4) * 8);
        const float* b2l = b2 + (size_t)layer * HC;
        const float* b3l = b3 + (size_t)layer * HKC;

        // ---- GEMM2: h2 = relu(h1 @ W2 + b2) --------------------------------
        for (int t = 0; t < 8; ++t) {
            v8f acc = {};
            for (int kk = 0; kk < 4; ++kk) {
                FragH a, b;
                const _Float16* ap = bufA + mrow * HC + kk * 32 + kgrpA;
                uint4 alo = *(const uint4*)ap;
                uint4 ahi = *(const uint4*)(ap + 16);
                a.u[0] = alo.x; a.u[1] = alo.y; a.u[2] = alo.z; a.u[3] = alo.w;
                a.u[4] = ahi.x; a.u[5] = ahi.y; a.u[6] = ahi.z; a.u[7] = ahi.w;
                const _Float16* bp = sW2 + (t * 16 + n15) * HC + kk * 32 + kgrpB;
                uint4 blo = *(const uint4*)bp;
                uint4 bhi = *(const uint4*)(bp + 16);
                b.u[0] = blo.x; b.u[1] = blo.y; b.u[2] = blo.z; b.u[3] = blo.w;
                b.u[4] = bhi.x; b.u[5] = bhi.y; b.u[6] = bhi.z; b.u[7] = bhi.w;
                acc = __builtin_amdgcn_wmma_f32_16x16x32_f16(
                    false, a.v, false, b.v, (short)0, acc, false, false);
            }
            const int nc   = t * 16 + n15;
            const float bias = b2l[nc];
#pragma unroll
            for (int r = 0; r < 8; ++r) {
                float vv = fmaxf(acc[r] + bias, 0.0f);
                bufB[(mbase + r) * HC + nc] = (_Float16)vv;
            }
        }

        // ---- GEMM3: theta = h2 @ W3 + b3 (no relu) -> bufA -----------------
        // No barrier: bufB rows produced and consumed by the same wave.
        for (int t = 0; t < 8; ++t) {
            v8f acc = {};
            for (int kk = 0; kk < 4; ++kk) {
                FragH a, b;
                const _Float16* ap = bufB + mrow * HC + kk * 32 + kgrpA;
                uint4 alo = *(const uint4*)ap;
                uint4 ahi = *(const uint4*)(ap + 16);
                a.u[0] = alo.x; a.u[1] = alo.y; a.u[2] = alo.z; a.u[3] = alo.w;
                a.u[4] = ahi.x; a.u[5] = ahi.y; a.u[6] = ahi.z; a.u[7] = ahi.w;
                const _Float16* bp = sW3 + (t * 16 + n15) * HC + kk * 32 + kgrpB;
                uint4 blo = *(const uint4*)bp;
                uint4 bhi = *(const uint4*)(bp + 16);
                b.u[0] = blo.x; b.u[1] = blo.y; b.u[2] = blo.z; b.u[3] = blo.w;
                b.u[4] = bhi.x; b.u[5] = bhi.y; b.u[6] = bhi.z; b.u[7] = bhi.w;
                acc = __builtin_amdgcn_wmma_f32_16x16x32_f16(
                    false, a.v, false, b.v, (short)0, acc, false, false);
            }
            const int nc   = t * 16 + n15;
            const float bias = (nc < HKC) ? b3l[nc] : 0.0f;
#pragma unroll
            for (int r = 0; r < 8; ++r) {
                bufA[(mbase + r) * HC + nc] = (_Float16)(acc[r] + bias);
            }
        }
        __syncthreads();

        // ---- linear spline per (sample, half-dim): softmax + cumsum --------
        for (int task = tid; task < BLK_S * HALFC; task += NTHR) {
            const int sl = task / HALFC;
            const int d  = task % HALFC;
            float xt  = xtb[task];
            float u   = fminf(fmaxf((xt + 1.0f) * 0.5f, 0.0f), 1.0f);
            float pos = u * (float)KC;
            int   idx = (int)floorf(pos);
            idx = idx < 0 ? 0 : (idx > KC - 1 ? KC - 1 : idx);
            float alpha = pos - (float)idx;
            // vectorized theta fetch: 5 x ds_load_b128 = 40 halves
            const uint4* thp = (const uint4*)(bufA + sl * HC + d * KC);
            float thf[KC];
            union { uint4 q; _Float16 h[8]; } cv;
#pragma unroll
            for (int g = 0; g < KC / 8; ++g) {
                cv.q = thp[g];
#pragma unroll
                for (int e = 0; e < 8; ++e) thf[g * 8 + e] = (float)cv.h[e];
            }
            float mx = -1e30f;
#pragma unroll
            for (int q = 0; q < KC; ++q) mx = fmaxf(mx, thf[q]);
            float sum = 0.0f, csum = 0.0f, pe = 0.0f;
#pragma unroll
            for (int q = 0; q < KC; ++q) {
                float e = __expf(thf[q] - mx);
                sum += e;
                if (q <= idx) csum += e;
                if (q == idx) pe = e;
            }
            float inv = 1.0f / sum;
            float p = pe * inv;
            float c = csum * inv;
            ytb[task] = (c - p + alpha * p) * 2.0f - 1.0f;
        }
        __syncthreads();

        // ---- scatter through argsort(order), then inter-layer perm ---------
        if (tid < BLK_S) {
            float yp[DIMC];
            yp[0] = yblk[tid * DIMC + ord[0]];
            yp[1] = yblk[tid * DIMC + ord[1]];
            yp[2] = yblk[tid * DIMC + ord[2]];
            yp[3] = ytb[tid * HALFC + 0];
            yp[4] = ytb[tid * HALFC + 1];
            yp[5] = ytb[tid * HALFC + 2];
            // take(yp, argsort(order))  <=>  y[order[d]] = yp[d]
#pragma unroll
            for (int d = 0; d < DIMC; ++d) yblk[tid * DIMC + ord[d]] = yp[d];
            float t6[DIMC];
#pragma unroll
            for (int j = 0; j < DIMC; ++j) t6[j] = yblk[tid * DIMC + prm[j]];
#pragma unroll
            for (int j = 0; j < DIMC; ++j) yblk[tid * DIMC + j] = t6[j];
        }
        __syncthreads();
    }

    // ---- write output ------------------------------------------------------
    if (tid < BLK_S) {
        const int s = s0 + tid;
#pragma unroll
        for (int j = 0; j < DIMC; ++j) out[s * DIMC + j] = yblk[tid * DIMC + j];
    }
}

// ---------------------------------------------------------------------------
extern "C" void kernel_launch(void* const* d_in, const int* in_sizes, int n_in,
                              void* d_out, int out_size, void* d_ws, size_t ws_size,
                              hipStream_t stream) {
    const float* x      = (const float*)d_in[0];
    const float* W1     = (const float*)d_in[1];
    const float* b1     = (const float*)d_in[2];
    const float* W2     = (const float*)d_in[3];
    const float* b2     = (const float*)d_in[4];
    const float* W3     = (const float*)d_in[5];
    const float* b3     = (const float*)d_in[6];
    const int*   orders = (const int*)d_in[7];
    const int*   perms  = (const int*)d_in[8];
    float*       out    = (float*)d_out;

    const int B = in_sizes[0] / DIMC;

    _Float16* W2t = (_Float16*)d_ws;
    _Float16* W3t = W2t + (size_t)NLC * HC * HC;

    const int totalW = NLC * HC * HC;
    se3flow_prep_weights<<<(totalW + 255) / 256, 256, 0, stream>>>(W2, W3, W2t, W3t);

    se3flow_kernel<<<B / BLK_S, NTHR, 0, stream>>>(
        x, W1, b1, b2, b3, orders, perms, W2t, W3t, out);
}
